// DeepSeekMoE_73658689126741
// MI455X (gfx1250) — matrix operand order repaired and checked by reading, not compiled, not run on hardware
//
#include <hip/hip_runtime.h>
#include <hip/hip_bf16.h>

// ---------------------------------------------------------------------------
// DeepSeek-style MoE forward for MI455X (gfx1250, wave32, WMMA).
// Pipeline: fp32->bf16 convert, gate GEMM (erf-GELU) via WMMA, per-token
// softmax/top-2, dispatch (count/scan/place/gather), shared-expert GEMMs,
// routed-expert GEMMs over compacted 128-aligned row ranges (single launch
// covering all experts), weighted combine.
// GEMM: 128x64 block tile, 8 waves x (32x32 wave tile) = 4 WMMA/K-step/wave,
// double-buffered LDS (one barrier per K-step), global prefetch of next tiles.
// ---------------------------------------------------------------------------

typedef __attribute__((ext_vector_type(16))) __bf16 v16bf;
typedef __attribute__((ext_vector_type(8)))  __bf16 v8bf;
typedef __attribute__((ext_vector_type(8)))  float  v8f;

constexpr int B_ = 4, L_ = 2048, D_ = 1024, H_ = 2048, E_ = 8, K_ = 2;
constexpr int T_ = B_ * L_;             // 8192 tokens
constexpr int S_ = K_ * T_ + E_ * 128;  // compact slot space, 128-aligned pads

enum { ACT_NONE = 0, ACT_GELU_TANH = 1, ACT_GELU_ERF = 2 };

__device__ __forceinline__ float gelu_tanh_f(float x) {
    float x3 = x * x * x;
    return 0.5f * x * (1.0f + tanhf(0.7978845608028654f * (x + 0.044715f * x3)));
}
__device__ __forceinline__ float gelu_erf_f(float x) {
    return 0.5f * x * (1.0f + erff(x * 0.7071067811865476f));
}

// ---------------------------------------------------------------------------
// fp32 -> bf16 conversion (4 elems / thread)
// ---------------------------------------------------------------------------
__global__ __launch_bounds__(256) void cvt_kernel(const float* __restrict__ in,
                                                  __bf16* __restrict__ out, size_t n) {
    size_t i = ((size_t)blockIdx.x * 256 + threadIdx.x) * 4;
    if (i < n) {
        float4 v = *(const float4*)(in + i);
        out[i + 0] = (__bf16)v.x;
        out[i + 1] = (__bf16)v.y;
        out[i + 2] = (__bf16)v.z;
        out[i + 3] = (__bf16)v.w;
    }
}

__global__ __launch_bounds__(256) void fill_zero_kernel(int* __restrict__ p, int n) {
    int i = blockIdx.x * 256 + threadIdx.x;
    if (i < n) p[i] = 0;
}

// ---------------------------------------------------------------------------
// Tiled WMMA GEMM: C[M,N] = act(A[M,K] @ W[K,N] + bias), bf16 A/W, f32 acc.
// Block tile 128x64, 8 waves (4 M x 2 N), each wave owns 32x32 -> 4 WMMAs per
// 32-deep K-step. LDS is double-buffered: one barrier per K-step.
// EXPERT mode: blockIdx.y row-block is mapped to an expert via 128-aligned
// `offsets[]`; W/bias are offset by the expert stride.
// ---------------------------------------------------------------------------
template <int ACT, typename TOUT, bool EXPERT>
__global__ __launch_bounds__(256) void gemm_kernel(
    const __bf16* __restrict__ A, const __bf16* __restrict__ Wb,
    const float* __restrict__ biasb, TOUT* __restrict__ C,
    int M, int N, int Kd, const int* __restrict__ offsets,
    long wstride, int bstride) {
    constexpr int BM = 128, BN = 64, BK = 32, PA = 40, PB = 40;
    __shared__ __bf16 As[2][BM * PA];   // 2 x 10240 B
    __shared__ __bf16 Bs[2][BN * PB];   // 2 x  5120 B  (stored transposed [n][k])

    const int r0 = blockIdx.y * BM;
    const __bf16* W = Wb;
    const float* bias = biasb;
    if constexpr (EXPERT) {
        if (r0 >= offsets[E_]) return;          // beyond total routed rows
        int e = 0;
        while (r0 >= offsets[e + 1]) ++e;       // 128-aligned: block in one expert
        W = Wb + (long)e * wstride;
        bias = biasb + (long)e * bstride;
    } else {
        if (r0 >= M) return;
    }
    const int n0 = blockIdx.x * BN;
    const int tid = threadIdx.x;
    const int lane = tid & 31;
    const int wv = tid >> 5;               // wave 0..7
    const int wm = (wv >> 1) * 32;         // wave M tile: 0/32/64/96
    const int wn = (wv & 1) * 32;          // wave N tile: 0/32
    const int hh = lane >> 4;              // lane half (ISA fragment layout)
    const int l15 = lane & 15;

    v8f acc00 = {}, acc01 = {}, acc10 = {}, acc11 = {};

    // cooperative tile loads: A 128x32 (16 elems/thread), B 32x64 (8 elems/thread)
    const int arow = tid >> 1, acol = (tid & 1) * 16;
    const int brow = tid >> 3, bcol = (tid & 7) * 8;

    const __bf16* gA = A + (size_t)(r0 + arow) * Kd + acol;
    const __bf16* gB = W + (size_t)brow * N + n0 + bcol;

    // prologue: stage K-tile 0 into buffer 0
    v8bf ra0 = *(const v8bf*)(gA);
    v8bf ra1 = *(const v8bf*)(gA + 8);
    v8bf rb  = *(const v8bf*)(gB);
    *(v8bf*)&As[0][arow * PA + acol] = ra0;
    *(v8bf*)&As[0][arow * PA + acol + 8] = ra1;
#pragma unroll
    for (int j = 0; j < 8; ++j) Bs[0][(bcol + j) * PB + brow] = rb[j];  // transpose
    __syncthreads();

    int buf = 0;
    for (int k0 = 0; k0 < Kd; k0 += BK) {
        const bool more = (k0 + BK) < Kd;
        if (more) {                          // global loads for next K-tile
            ra0 = *(const v8bf*)(gA + k0 + BK);
            ra1 = *(const v8bf*)(gA + k0 + BK + 8);
            rb  = *(const v8bf*)(gB + (size_t)(k0 + BK) * N);
            if (k0 + 2 * BK < Kd) {
                __builtin_prefetch(gA + k0 + 2 * BK, 0, 1);
                __builtin_prefetch(gB + (size_t)(k0 + 2 * BK) * N, 0, 1);
            }
        }

        // A fragments (16x32 bf16): lanes<16 hold K {0..7,16..23}, lanes>=16
        // hold K {8..15,24..31} for row M=lane&15 (ISA 7.12.2).
        v16bf af0, af1, bf0, bf1;
        const __bf16* ap0 = &As[buf][(wm + l15) * PA + hh * 8];
        const __bf16* ap1 = ap0 + 16 * PA;
#pragma unroll
        for (int i = 0; i < 8; ++i) {
            af0[i] = ap0[i]; af0[8 + i] = ap0[16 + i];
            af1[i] = ap1[i]; af1[8 + i] = ap1[16 + i];
        }
        // B fragments (32x16): lanes<16 hold K {0..7,16..23}, lanes>=16 hold
        // K {8..15,24..31} for column N=lane&15; Bs is stored [n][k].
        const __bf16* bp0 = &Bs[buf][(wn + l15) * PB + hh * 8];
        const __bf16* bp1 = bp0 + 16 * PB;
#pragma unroll
        for (int i = 0; i < 8; ++i) {
            bf0[i] = bp0[i]; bf0[8 + i] = bp0[16 + i];
            bf1[i] = bp1[i]; bf1[8 + i] = bp1[16 + i];
        }

        acc00 = __builtin_amdgcn_wmma_f32_16x16x32_bf16(false, af0, false, bf0,
                                                        (short)0, acc00, false, false);
        acc01 = __builtin_amdgcn_wmma_f32_16x16x32_bf16(false, af0, false, bf1,
                                                        (short)0, acc01, false, false);
        acc10 = __builtin_amdgcn_wmma_f32_16x16x32_bf16(false, af1, false, bf0,
                                                        (short)0, acc10, false, false);
        acc11 = __builtin_amdgcn_wmma_f32_16x16x32_bf16(false, af1, false, bf1,
                                                        (short)0, acc11, false, false);

        if (more) {                          // stage next tile into other buffer
            const int nb = buf ^ 1;
            *(v8bf*)&As[nb][arow * PA + acol] = ra0;
            *(v8bf*)&As[nb][arow * PA + acol + 8] = ra1;
#pragma unroll
            for (int j = 0; j < 8; ++j) Bs[nb][(bcol + j) * PB + brow] = rb[j];
        }
        buf ^= 1;
        __syncthreads();                     // single barrier per K-step
    }

    // C/D layout: lanes<16 col=lane rows +0..7; lanes>=16 col=lane-16 rows +8..15
    const int crow0 = r0 + wm + hh * 8;          // acc0x (M sub-tile 0)
    const int crow1 = crow0 + 16;                // acc1x (M sub-tile 1)
    const int c0 = n0 + wn + l15;
    const int c1 = c0 + 16;
    const float b0 = bias[c0], b1 = bias[c1];
#pragma unroll
    for (int r = 0; r < 8; ++r) {
        float v00 = acc00[r] + b0, v01 = acc01[r] + b1;
        float v10 = acc10[r] + b0, v11 = acc11[r] + b1;
        if (ACT == ACT_GELU_TANH) {
            v00 = gelu_tanh_f(v00); v01 = gelu_tanh_f(v01);
            v10 = gelu_tanh_f(v10); v11 = gelu_tanh_f(v11);
        } else if (ACT == ACT_GELU_ERF) {
            v00 = gelu_erf_f(v00); v01 = gelu_erf_f(v01);
            v10 = gelu_erf_f(v10); v11 = gelu_erf_f(v11);
        }
        C[(size_t)(crow0 + r) * N + c0] = (TOUT)v00;
        C[(size_t)(crow0 + r) * N + c1] = (TOUT)v01;
        C[(size_t)(crow1 + r) * N + c0] = (TOUT)v10;
        C[(size_t)(crow1 + r) * N + c1] = (TOUT)v11;
    }
}

// ---------------------------------------------------------------------------
// Gating: one block per token; wave e computes logit_e = <hid[t], gw2[:,e]>,
// thread 0 does softmax + top-2, records expert ids/weights, bumps counts.
// ---------------------------------------------------------------------------
__global__ __launch_bounds__(256) void gate_topk_kernel(
    const __bf16* __restrict__ hid, const float* __restrict__ gw2,
    const float* __restrict__ gb2, int* __restrict__ texp,
    float* __restrict__ twgt, int* __restrict__ counts) {
    const int t = blockIdx.x;
    const int lane = threadIdx.x & 31;
    const int e = threadIdx.x >> 5;  // 8 waves -> 8 experts
    const __bf16* hrow = hid + (size_t)t * H_;
    float s = 0.0f;
    for (int h = lane; h < H_; h += 32) s += (float)hrow[h] * gw2[h * E_ + e];
#pragma unroll
    for (int o = 16; o > 0; o >>= 1) s += __shfl_xor(s, o, 32);
    __shared__ float lg[E_];
    if (lane == 0) lg[e] = s + gb2[e];
    __syncthreads();
    if (threadIdx.x == 0) {
        float m = lg[0];
#pragma unroll
        for (int i = 1; i < E_; ++i) m = fmaxf(m, lg[i]);
        float p[E_], sum = 0.0f;
#pragma unroll
        for (int i = 0; i < E_; ++i) { p[i] = expf(lg[i] - m); sum += p[i]; }
        float inv = 1.0f / sum;
        int i1 = 0;
#pragma unroll
        for (int i = 1; i < E_; ++i) if (p[i] > p[i1]) i1 = i;
        int i2 = (i1 == 0) ? 1 : 0;
#pragma unroll
        for (int i = 0; i < E_; ++i) if (i != i1 && p[i] > p[i2]) i2 = i;
        texp[t * 2 + 0] = i1; twgt[t * 2 + 0] = p[i1] * inv;
        texp[t * 2 + 1] = i2; twgt[t * 2 + 1] = p[i2] * inv;
        atomicAdd(&counts[i1], 1);
        atomicAdd(&counts[i2], 1);
    }
}

__global__ void scan_kernel(const int* __restrict__ counts, int* __restrict__ offsets) {
    if (blockIdx.x == 0 && threadIdx.x == 0) {
        int acc = 0;
        for (int e = 0; e < E_; ++e) {
            offsets[e] = acc;
            acc += (counts[e] + 127) & ~127;  // 128-align each expert range
        }
        offsets[E_] = acc;
    }
}

__global__ __launch_bounds__(256) void place_kernel(
    const int* __restrict__ texp, const int* __restrict__ offsets,
    int* __restrict__ counts2, int* __restrict__ list, int* __restrict__ slot_of) {
    const int t = blockIdx.x * 256 + threadIdx.x;
    if (t >= T_) return;
#pragma unroll
    for (int k = 0; k < K_; ++k) {
        int e = texp[t * 2 + k];
        int pos = atomicAdd(&counts2[e], 1);
        int slot = offsets[e] + pos;
        list[slot] = t;
        slot_of[t * 2 + k] = slot;
    }
}

__global__ __launch_bounds__(256) void gather_x_kernel(
    const __bf16* __restrict__ xb, const int* __restrict__ list,
    __bf16* __restrict__ xg) {
    size_t i = (size_t)blockIdx.x * 256 + threadIdx.x;  // one 8-elem chunk
    int slot = (int)(i / (D_ / 8));
    int c = (int)(i % (D_ / 8)) * 8;
    if (slot >= S_) return;
    int tok = list[slot];  // pad slots hold 0 -> benign token-0 row
    *(v8bf*)(xg + (size_t)slot * D_ + c) = *(const v8bf*)(xb + (size_t)tok * D_ + c);
}

__global__ __launch_bounds__(256) void combine_kernel(
    const float* __restrict__ yout, const int* __restrict__ slot_of,
    const float* __restrict__ twgt, float* __restrict__ out) {
    const int t = blockIdx.x;
    const int s0 = slot_of[t * 2 + 0], s1 = slot_of[t * 2 + 1];
    const float w0 = twgt[t * 2 + 0], w1 = twgt[t * 2 + 1];
    const float* y0 = yout + (size_t)s0 * D_;
    const float* y1 = yout + (size_t)s1 * D_;
    float* o = out + (size_t)t * D_;
    for (int d = threadIdx.x; d < D_; d += 256)
        o[d] += w0 * y0[d] + w1 * y1[d];
}

// ---------------------------------------------------------------------------
extern "C" void kernel_launch(void* const* d_in, const int* in_sizes, int n_in,
                              void* d_out, int out_size, void* d_ws, size_t ws_size,
                              hipStream_t stream) {
    const float* x   = (const float*)d_in[0];
    // d_in[1] = task_ids (unused by reference math)
    const float* gw1 = (const float*)d_in[2];
    const float* gb1 = (const float*)d_in[3];
    const float* gw2 = (const float*)d_in[4];
    const float* gb2 = (const float*)d_in[5];
    const float* We1 = (const float*)d_in[6];
    const float* be1 = (const float*)d_in[7];
    const float* We2 = (const float*)d_in[8];
    const float* be2 = (const float*)d_in[9];
    const float* Ws1 = (const float*)d_in[10];
    const float* bs1 = (const float*)d_in[11];
    const float* Ws2 = (const float*)d_in[12];
    const float* bs2 = (const float*)d_in[13];
    float* out = (float*)d_out;

    char* p = (char*)d_ws;
    auto take = [&](size_t bytes) -> char* {
        char* r = p;
        p += (bytes + 255) & ~(size_t)255;
        return r;
    };
    __bf16* xb   = (__bf16*)take((size_t)T_ * D_ * 2);
    __bf16* gw1b = (__bf16*)take((size_t)D_ * H_ * 2);
    __bf16* ws1b = (__bf16*)take((size_t)D_ * H_ * 2);
    __bf16* ws2b = (__bf16*)take((size_t)H_ * D_ * 2);
    __bf16* we1b = (__bf16*)take((size_t)E_ * D_ * H_ * 2);
    __bf16* we2b = (__bf16*)take((size_t)E_ * H_ * D_ * 2);
    __bf16* hid  = (__bf16*)take((size_t)T_ * H_ * 2);   // gate hidden, then shared hidden
    __bf16* xg   = (__bf16*)take((size_t)S_ * D_ * 2);   // gathered routed inputs
    __bf16* ehid = (__bf16*)take((size_t)S_ * H_ * 2);   // routed hidden
    float*  yout = (float*) take((size_t)S_ * D_ * 4);   // routed outputs per slot
    int* meta    = (int*)   take((size_t)(32 + S_) * 4); // counts|counts2|offsets|list
    int* counts  = meta;
    int* counts2 = meta + 8;
    int* offsets = meta + 16;  // E_+1 = 9 used
    int* list    = meta + 32;
    int* texp    = (int*)  take((size_t)T_ * 2 * 4);
    float* twgt  = (float*)take((size_t)T_ * 2 * 4);
    int* slot_of = (int*)  take((size_t)T_ * 2 * 4);

    const dim3 blk(256);

    // fp32 -> bf16 conversions (x + all GEMM weights)
    cvt_kernel<<<(T_ * (size_t)D_) / 1024, blk, 0, stream>>>(x, xb, (size_t)T_ * D_);
    cvt_kernel<<<((size_t)D_ * H_) / 1024, blk, 0, stream>>>(gw1, gw1b, (size_t)D_ * H_);
    cvt_kernel<<<((size_t)D_ * H_) / 1024, blk, 0, stream>>>(Ws1, ws1b, (size_t)D_ * H_);
    cvt_kernel<<<((size_t)H_ * D_) / 1024, blk, 0, stream>>>(Ws2, ws2b, (size_t)H_ * D_);
    cvt_kernel<<<((size_t)E_ * D_ * H_) / 1024, blk, 0, stream>>>(We1, we1b, (size_t)E_ * D_ * H_);
    cvt_kernel<<<((size_t)E_ * H_ * D_) / 1024, blk, 0, stream>>>(We2, we2b, (size_t)E_ * H_ * D_);

    // zero per-call dispatch state (counts, counts2, offsets, list)
    fill_zero_kernel<<<(32 + S_ + 255) / 256, blk, 0, stream>>>(meta, 32 + S_);

    // gate: hid = gelu_erf(x @ gw1 + gb1)  [T,H]
    gemm_kernel<ACT_GELU_ERF, __bf16, false><<<dim3(H_ / 64, T_ / 128), blk, 0, stream>>>(
        xb, gw1b, gb1, hid, T_, H_, D_, nullptr, 0, 0);
    // logits/softmax/top-2 per token
    gate_topk_kernel<<<T_, blk, 0, stream>>>(hid, gw2, gb2, texp, twgt, counts);
    scan_kernel<<<1, 32, 0, stream>>>(counts, offsets);
    place_kernel<<<T_ / 256, blk, 0, stream>>>(texp, offsets, counts2, list, slot_of);
    gather_x_kernel<<<(S_ * (D_ / 8)) / 256, blk, 0, stream>>>(xb, list, xg);

    // shared expert: out = gelu_tanh(x @ Ws1 + bs1) @ Ws2 + bs2   (overwrites d_out)
    gemm_kernel<ACT_GELU_TANH, __bf16, false><<<dim3(H_ / 64, T_ / 128), blk, 0, stream>>>(
        xb, ws1b, bs1, hid, T_, H_, D_, nullptr, 0, 0);
    gemm_kernel<ACT_NONE, float, false><<<dim3(D_ / 64, T_ / 128), blk, 0, stream>>>(
        hid, ws2b, bs2, out, T_, D_, H_, nullptr, 0, 0);

    // routed experts over compacted 128-aligned row ranges (one launch each)
    gemm_kernel<ACT_GELU_TANH, __bf16, true><<<dim3(H_ / 64, S_ / 128), blk, 0, stream>>>(
        xg, we1b, be1, ehid, S_, H_, D_, offsets, (long)D_ * H_, H_);
    gemm_kernel<ACT_NONE, float, true><<<dim3(D_ / 64, S_ / 128), blk, 0, stream>>>(
        ehid, we2b, be2, yout, S_, D_, H_, offsets, (long)H_ * D_, D_);

    // out[t] += w0 * y[slot0(t)] + w1 * y[slot1(t)]
    combine_kernel<<<T_, blk, 0, stream>>>(yout, slot_of, twgt, out);
}
